// Multi_Head_Attention_25091198943737
// MI455X (gfx1250) — matrix operand order repaired and checked by reading, not compiled, not run on hardware
//
#include <hip/hip_runtime.h>
#include <stdint.h>

// ---------------- constants ----------------
#define BATCH   2
#define SEQ     2048
#define DMODEL  1024
#define NHEAD   16
#define DHEAD   64
#define MTOT    (BATCH * SEQ)          // 4096 rows for projections

typedef __attribute__((ext_vector_type(16))) __bf16 v16bf;
typedef __attribute__((ext_vector_type(8)))  float  v8f;

#define DEV static __device__ __forceinline__

DEV unsigned short f2bf(float f) {
  union { float f; unsigned u; } a; a.f = f;
  unsigned u = a.u;
  unsigned r = (u + 0x7FFFu + ((u >> 16) & 1u)) >> 16;   // round-to-nearest-even
  return (unsigned short)r;
}

// build a v16bf fragment from two 16-byte chunks (possibly non-contiguous)
DEV v16bf ldv16(const unsigned short* p0, const unsigned short* p1) {
  union { v16bf v; uint4 q[2]; } u;
  u.q[0] = *(const uint4*)p0;
  u.q[1] = *(const uint4*)p1;
  return u.v;
}

DEV v8f wmma_bf16(v16bf a, v16bf b, v8f c) {
  // (neg_a, A, neg_b, B, c_mod, C, reuse_a, reuse_b)
  return __builtin_amdgcn_wmma_f32_16x16x32_bf16(false, a, false, b, (short)0, c,
                                                 false, false);
}

// async global -> LDS copy of 16 bytes, tracked with ASYNCcnt.
// Operands: VDST = VGPR with LDS byte address (generic shared ptr low 32 bits),
// VADDR = 64-bit global address, saddr = off.
DEV void async_copy16(const unsigned short* g, unsigned short* l) {
  unsigned loff = (unsigned)(unsigned long long)(uintptr_t)l;   // LDS byte offset
  unsigned long long ga = (unsigned long long)(uintptr_t)g;
  asm volatile("global_load_async_to_lds_b128 %0, %1, off"
               :: "v"(loff), "v"(ga) : "memory");
}

DEV void wait_async0() { asm volatile("s_wait_asynccnt 0x0" ::: "memory"); }

// ---------------- kernel 1: fp32 -> bf16 convert ----------------
__global__ __launch_bounds__(256) void cvt_bf16_kernel(const float4* __restrict__ src,
                                                       ushort4* __restrict__ dst, int n4) {
  int i = blockIdx.x * blockDim.x + threadIdx.x;
  if (i < n4) {
    float4 f = src[i];
    ushort4 o;
    o.x = f2bf(f.x); o.y = f2bf(f.y); o.z = f2bf(f.z); o.w = f2bf(f.w);
    dst[i] = o;
  }
}

// ---------------- kernel 2: projection GEMM  P = X @ W^T + b ----------------
// X: [MTOT, 1024] bf16 row-major, W: [1024, 1024] bf16 row-major (W[n][k])
// mode 0 (Q): store -(v)/8        as [B,H,S,64]
// mode 1 (K): store  v            as [B,H,S,64]
// mode 2 (V): store  v transposed as [B,H,64,S]
__global__ __launch_bounds__(128) void proj_kernel(const unsigned short* __restrict__ X,
                                                   const unsigned short* __restrict__ W,
                                                   const float* __restrict__ bias,
                                                   unsigned short* __restrict__ out,
                                                   int mode) {
  const int tid = threadIdx.x;
  const int wid = tid >> 5;
  const int lane = tid & 31;
  const int lane16 = lane & 15;
  const int kh = lane >> 4;                    // K-half selector for fragments

  const int m0 = blockIdx.x * 64 + (wid >> 1) * 32;
  const int n0 = blockIdx.y * 64 + (wid & 1) * 32;

  v8f acc[2][2];
#pragma unroll
  for (int i = 0; i < 2; ++i)
#pragma unroll
    for (int j = 0; j < 2; ++j)
#pragma unroll
      for (int r = 0; r < 8; ++r) acc[i][j][r] = 0.0f;

  const unsigned short* Arow[2];
  const unsigned short* Brow[2];
#pragma unroll
  for (int i = 0; i < 2; ++i) {
    Arow[i] = X + (size_t)(m0 + i * 16 + lane16) * DMODEL;
    Brow[i] = W + (size_t)(n0 + i * 16 + lane16) * DMODEL;
  }

  for (int k0 = 0; k0 < DMODEL; k0 += 32) {
    // A fragment (16x32): lane<16 -> K {0..7,16..23}; lane>=16 -> K {8..15,24..31}
    v16bf a0 = ldv16(Arow[0] + k0 + kh * 8, Arow[0] + k0 + 16 + kh * 8);
    v16bf a1 = ldv16(Arow[1] + k0 + kh * 8, Arow[1] + k0 + 16 + kh * 8);
    // B fragment (32x16): lane = N col, K {0..7,16..23} / {8..15,24..31} like A
    v16bf b0 = ldv16(Brow[0] + k0 + kh * 8, Brow[0] + k0 + 16 + kh * 8);
    v16bf b1 = ldv16(Brow[1] + k0 + kh * 8, Brow[1] + k0 + 16 + kh * 8);
    acc[0][0] = wmma_bf16(a0, b0, acc[0][0]);
    acc[0][1] = wmma_bf16(a0, b1, acc[0][1]);
    acc[1][0] = wmma_bf16(a1, b0, acc[1][0]);
    acc[1][1] = wmma_bf16(a1, b1, acc[1][1]);
  }

  const float bv0 = bias[n0 + lane16];
  const float bv1 = bias[n0 + 16 + lane16];

#pragma unroll
  for (int mi = 0; mi < 2; ++mi)
#pragma unroll
    for (int nj = 0; nj < 2; ++nj)
#pragma unroll
      for (int r = 0; r < 8; ++r) {
        int m = m0 + mi * 16 + r + 8 * kh;       // C/D layout: VGPR r -> row r (+8 hi half)
        int n = n0 + nj * 16 + lane16;
        float v = acc[mi][nj][r] + (nj ? bv1 : bv0);
        int b = m >> 11, s = m & 2047;
        int h = n >> 6,  d = n & 63;
        size_t idx;
        if (mode == 2) {                         // V transposed: [B,H,64,S]
          idx = (((size_t)(b * NHEAD + h) * DHEAD + d) * SEQ) + s;
        } else {                                 // Q/K: [B,H,S,64]
          idx = (((size_t)(b * NHEAD + h) * SEQ + s) * DHEAD) + d;
          if (mode == 0) v = -0.125f * v;        // fold -1/sqrt(d) into Q
        }
        out[idx] = f2bf(v);
      }
}

// stage one 32-key K tile (32x64 bf16) and V tile (64x32 bf16) into LDS
// cooperatively: 128 threads x 32B each per tile, via async data mover.
DEV void stage_tiles(const unsigned short* Kb, const unsigned short* Vb, int k0,
                     unsigned short* kl, unsigned short* vl, int tid) {
  const unsigned short* ks = Kb + (size_t)k0 * DHEAD + tid * 16;  // K tile contiguous 4KB
  unsigned short* kd = kl + tid * 16;
  async_copy16(ks, kd);
  async_copy16(ks + 8, kd + 8);
  const int d = tid >> 1, half = tid & 1;                         // Vt rows strided by SEQ
  const unsigned short* vs = Vb + (size_t)d * SEQ + k0 + half * 16;
  unsigned short* vd = vl + d * 32 + half * 16;
  async_copy16(vs, vd);
  async_copy16(vs + 8, vd + 8);
}

// ---------------- kernel 3: flash attention ----------------
// Q:[B,H,S,64] (pre-scaled -Q/8), K:[B,H,S,64], Vt:[B,H,64,S], all bf16
// mask:[B,S,S] fp32, out:[B,S,H*64] fp32
__global__ __launch_bounds__(128) void attn_kernel(const unsigned short* __restrict__ Q,
                                                   const unsigned short* __restrict__ K,
                                                   const unsigned short* __restrict__ Vt,
                                                   const float* __restrict__ mask,
                                                   float* __restrict__ out) {
  __shared__ unsigned short Klds[2][32 * DHEAD];   // [key_local][d]
  __shared__ unsigned short Vlds[2][DHEAD * 32];   // [d_local][key]
  __shared__ float Pbuf[4][16 * 32];

  const int tid = threadIdx.x;
  const int wid = tid >> 5;
  const int lane = tid & 31;
  const int lane16 = lane & 15;
  const int kh = lane >> 4;

  const int bh = blockIdx.y;
  const int b = bh >> 4;
  const int h = bh & 15;
  const int q0 = blockIdx.x * 64 + wid * 16;

  const unsigned short* Qb = Q + (size_t)bh * SEQ * DHEAD;
  const unsigned short* Kb = K + (size_t)bh * SEQ * DHEAD;
  const unsigned short* Vb = Vt + (size_t)bh * DHEAD * SEQ;
  const float* Mb = mask + (size_t)b * SEQ * SEQ;

  // Q A-fragments (reused across the whole key loop): rows q0..q0+15, K=d
  const unsigned short* qrow = Qb + (size_t)(q0 + lane16) * DHEAD;
  v16bf aq0 = ldv16(qrow + 0  + kh * 8, qrow + 16 + kh * 8);   // d 0..31
  v16bf aq1 = ldv16(qrow + 32 + kh * 8, qrow + 48 + kh * 8);   // d 32..63

  float mrow[8], lrow[8];
  v8f o[4];
#pragma unroll
  for (int r = 0; r < 8; ++r) { mrow[r] = -3.0e38f; lrow[r] = 0.0f; }
#pragma unroll
  for (int t = 0; t < 4; ++t)
#pragma unroll
    for (int r = 0; r < 8; ++r) o[t][r] = 0.0f;

  float* pb = Pbuf[wid];

  // prologue: async-stage tile 0
  stage_tiles(Kb, Vb, 0, Klds[0], Vlds[0], tid);

  for (int i = 0; i < SEQ / 32; ++i) {
    const int k0 = i * 32;
    const int cur = i & 1;

    wait_async0();        // my copies of tile i have landed in LDS
    __syncthreads();      // everyone's copies landed; prev buffer fully consumed

    // overlap: stream tile i+1 into the other buffer while computing tile i
    if (i + 1 < SEQ / 32)
      stage_tiles(Kb, Vb, k0 + 32, Klds[cur ^ 1], Vlds[cur ^ 1], tid);

    // mask tile loaded directly into the WMMA C operand (C/D layout)
    v8f s0, s1;
#pragma unroll
    for (int r = 0; r < 8; ++r) {
      const float* mr = Mb + (size_t)(q0 + r + 8 * kh) * SEQ + k0;
      s0[r] = mr[lane16];
      s1[r] = mr[16 + lane16];
    }
    if (k0 + 32 < SEQ)  // prefetch next mask rows (global_prefetch_b8)
      __builtin_prefetch(Mb + (size_t)(q0 + 8 * kh) * SEQ + k0 + 32, 0, 0);

    // K B-fragments from LDS: lane = key col, d in table order matching A
    const unsigned short* kr0 = &Klds[cur][lane16 * DHEAD];
    const unsigned short* kr1 = &Klds[cur][(16 + lane16) * DHEAD];
    v16bf bk00 = ldv16(kr0 + kh * 8,      kr0 + 16 + kh * 8);       // d 0..31
    v16bf bk01 = ldv16(kr0 + 32 + kh * 8, kr0 + 48 + kh * 8);  // d 32..63
    v16bf bk10 = ldv16(kr1 + kh * 8,      kr1 + 16 + kh * 8);
    v16bf bk11 = ldv16(kr1 + 32 + kh * 8, kr1 + 48 + kh * 8);

    s0 = wmma_bf16(aq0, bk00, s0);
    s0 = wmma_bf16(aq1, bk01, s0);
    s1 = wmma_bf16(aq0, bk10, s1);
    s1 = wmma_bf16(aq1, bk11, s1);

    // online softmax: rows span 16 lanes per wave half -> width-16 reductions
#pragma unroll
    for (int r = 0; r < 8; ++r) {
      float v = fmaxf(s0[r], s1[r]);
      v = fmaxf(v, __shfl_xor(v, 1, 16));
      v = fmaxf(v, __shfl_xor(v, 2, 16));
      v = fmaxf(v, __shfl_xor(v, 4, 16));
      v = fmaxf(v, __shfl_xor(v, 8, 16));
      float mn = fmaxf(mrow[r], v);
      float alpha = __expf(mrow[r] - mn);
      float p0 = __expf(s0[r] - mn);
      float p1 = __expf(s1[r] - mn);
      s0[r] = p0; s1[r] = p1;
      float rs = p0 + p1;
      rs += __shfl_xor(rs, 1, 16);
      rs += __shfl_xor(rs, 2, 16);
      rs += __shfl_xor(rs, 4, 16);
      rs += __shfl_xor(rs, 8, 16);
      lrow[r] = lrow[r] * alpha + rs;
      mrow[r] = mn;
#pragma unroll
      for (int t = 0; t < 4; ++t) o[t][r] *= alpha;
    }

    // bounce P through per-wave LDS scratch: C/D layout -> A layout (16x32)
#pragma unroll
    for (int r = 0; r < 8; ++r) {
      pb[(r + 8 * kh) * 32 + lane16]      = s0[r];
      pb[(r + 8 * kh) * 32 + 16 + lane16] = s1[r];
    }
    asm volatile("s_wait_dscnt 0" ::: "memory");

    union { v16bf v; unsigned short s[16]; } ap;
    const float* prow = pb + lane16 * 32;
#pragma unroll
    for (int j = 0; j < 8; ++j) {
      ap.s[j]     = f2bf(prow[kh * 8 + j]);        // K 0..7 / 8..15
      ap.s[8 + j] = f2bf(prow[16 + kh * 8 + j]);   // K 16..23 / 24..31
    }

    // V B-fragments from LDS Vt tile: lane = d col, keys in table order matching A
#pragma unroll
    for (int t = 0; t < 4; ++t) {
      const unsigned short* vp = &Vlds[cur][(t * 16 + lane16) * 32 + kh * 8];
      v16bf bv = ldv16(vp, vp + 16);
      o[t] = wmma_bf16(ap.v, bv, o[t]);
    }
  }

  // epilogue: normalize and write ctx -> [B, S, H*d]
#pragma unroll
  for (int t = 0; t < 4; ++t)
#pragma unroll
    for (int r = 0; r < 8; ++r) {
      int s = q0 + r + 8 * kh;
      float val = o[t][r] / lrow[r];
      out[((size_t)(b * SEQ + s)) * DMODEL + h * DHEAD + t * 16 + lane16] = val;
    }
}

// ---------------- host ----------------
extern "C" void kernel_launch(void* const* d_in, const int* in_sizes, int n_in,
                              void* d_out, int out_size, void* d_ws, size_t ws_size,
                              hipStream_t stream) {
  const float* in_Q  = (const float*)d_in[0];
  const float* in_K  = (const float*)d_in[1];
  const float* in_V  = (const float*)d_in[2];
  const float* mask  = (const float*)d_in[3];
  const float* Wq    = (const float*)d_in[4];
  const float* bq    = (const float*)d_in[5];
  const float* Wk    = (const float*)d_in[6];
  const float* bk    = (const float*)d_in[7];
  const float* Wv    = (const float*)d_in[8];
  const float* bv    = (const float*)d_in[9];
  float* out = (float*)d_out;

  const size_t NX = (size_t)MTOT * DMODEL;      // 4,194,304
  const size_t NW = (size_t)DMODEL * DMODEL;    // 1,048,576

  unsigned short* us = (unsigned short*)d_ws;
  unsigned short* Xq = us;            us += NX;
  unsigned short* Xk = us;            us += NX;
  unsigned short* Xv = us;            us += NX;
  unsigned short* Wqb = us;           us += NW;
  unsigned short* Wkb = us;           us += NW;
  unsigned short* Wvb = us;           us += NW;
  unsigned short* Qw  = us;           us += NX;
  unsigned short* Kw  = us;           us += NX;
  unsigned short* Vtw = us;           us += NX;

  // 1) convert fp32 -> bf16
  {
    int nx4 = (int)(NX / 4), nw4 = (int)(NW / 4);
    int bx = (nx4 + 255) / 256, bw = (nw4 + 255) / 256;
    cvt_bf16_kernel<<<bx, 256, 0, stream>>>((const float4*)in_Q, (ushort4*)Xq, nx4);
    cvt_bf16_kernel<<<bx, 256, 0, stream>>>((const float4*)in_K, (ushort4*)Xk, nx4);
    cvt_bf16_kernel<<<bx, 256, 0, stream>>>((const float4*)in_V, (ushort4*)Xv, nx4);
    cvt_bf16_kernel<<<bw, 256, 0, stream>>>((const float4*)Wq, (ushort4*)Wqb, nw4);
    cvt_bf16_kernel<<<bw, 256, 0, stream>>>((const float4*)Wk, (ushort4*)Wkb, nw4);
    cvt_bf16_kernel<<<bw, 256, 0, stream>>>((const float4*)Wv, (ushort4*)Wvb, nw4);
  }

  // 2) projections (WMMA GEMM)
  {
    dim3 grid(MTOT / 64, DMODEL / 64);
    proj_kernel<<<grid, 128, 0, stream>>>(Xq, Wqb, bq, Qw, 0);   // Q: -x/8
    proj_kernel<<<grid, 128, 0, stream>>>(Xk, Wkb, bk, Kw, 1);
    proj_kernel<<<grid, 128, 0, stream>>>(Xv, Wvb, bv, Vtw, 2);  // V transposed
  }

  // 3) fused flash attention (async double-buffered K/V tiles in LDS)
  {
    dim3 grid(SEQ / 64, BATCH * NHEAD);
    attn_kernel<<<grid, 128, 0, stream>>>(Qw, Kw, Vtw, mask, out);
  }
}
